// BilinearSeqAttnMix_11175504904495
// MI455X (gfx1250) — compile-verified, matrix operand order checked by reading, not run on hardware
//
#include <hip/hip_runtime.h>
#include <hip/hip_bf16.h>

#define BB 32
#define DD 2048
#define QQ 512
#define HH 1024

typedef __bf16 bf16_t;
typedef bf16_t v16bf __attribute__((ext_vector_type(16)));
typedef float  v8f   __attribute__((ext_vector_type(8)));
typedef unsigned int v4u __attribute__((ext_vector_type(4)));
typedef unsigned int v8u __attribute__((ext_vector_type(8)));

#define NEG_INF (-__builtin_inff())

static __device__ __forceinline__ bf16_t f2bf(float f) {
  unsigned u = __float_as_uint(f);
  unsigned r = (u + 0x7FFFu + ((u >> 16) & 1u)) >> 16;
  unsigned short s = (unsigned short)r;
  bf16_t o;
  __builtin_memcpy(&o, &s, 2);
  return o;
}
static __device__ __forceinline__ float bf2f(bf16_t b) {
  unsigned short s;
  __builtin_memcpy(&s, &b, 2);
  return __uint_as_float(((unsigned)s) << 16);
}
// 16-bit A/B fragment K-index map per CDNA5 ISA 7.12.2 (16-bit A 16x32 table)
static __device__ __forceinline__ int kmap(int e, int half8) {
  int v = e >> 1, lo = e & 1;
  return (v < 4) ? (2 * v + lo + half8) : (16 + 2 * (v - 4) + lo + half8);
}

// ---------------------------------------------------------------------------
// TDM: DMA a 2D bf16 tile (tile_rows x tile_k elems, row stride stride_elems)
// from global into LDS at byte offset lds_addr. Descriptor per ISA sec 8.3/8.4:
// group0 = {count=1 | lds_addr | global_addr(57b) | type=2},
// group1 = {mask=0,data_size=1(2B) | tensor dims | tile dims | dim0 stride}.
// Issued by one wave; completion via TENSORcnt.
// ---------------------------------------------------------------------------
static __device__ __forceinline__ void tdm_load_tile_bf16(
    unsigned lds_addr, const bf16_t* gptr, unsigned tile_k, unsigned tile_rows,
    unsigned tensor_d0, unsigned tensor_d1, unsigned stride_elems) {
  unsigned long long ga = (unsigned long long)(uintptr_t)gptr;
  v4u g0;
  g0[0] = 1u;                                                  // count=1 (user)
  g0[1] = lds_addr;                                            // lds_addr
  g0[2] = (unsigned)ga;                                        // global[31:0]
  g0[3] = (unsigned)((ga >> 32) & 0x01FFFFFFu) | (2u << 30);   // global[56:32]|type=2
  v8u g1;
  g1[0] = (1u << 16);                                          // data_size=1 (2B)
  g1[1] = (tensor_d0 & 0xFFFFu) << 16;                         // tensor_dim0[15:0]
  g1[2] = ((tensor_d0 >> 16) & 0xFFFFu) | ((tensor_d1 & 0xFFFFu) << 16);
  g1[3] = ((tensor_d1 >> 16) & 0xFFFFu) | (tile_k << 16);      // tile_dim0
  g1[4] = tile_rows;                                           // tile_dim1
  g1[5] = stride_elems;                                        // dim0_stride[31:0]
  g1[6] = 0u;
  g1[7] = 0u;
  asm volatile("tensor_load_to_lds %0, %1" ::"s"(g0), "s"(g1) : "memory");
}

// ---------------------------------------------------------------------------
// Kernel 1: y_n = tanh(y @ W1^T + b1)  [B*Q, H] x [H, H], output bf16
// Block tile 128(M) x 64(N); 8 waves = 4(M) x 2(N); each wave 4x 16x16 WMMA.
// ---------------------------------------------------------------------------
__global__ __launch_bounds__(256) void yn_gemm_tanh(
    const float* __restrict__ y, const float* __restrict__ W1,
    const float* __restrict__ b1, bf16_t* __restrict__ yn) {
  __shared__ bf16_t lA[128][32];
  __shared__ bf16_t lB[64][32];
  const int t = threadIdx.x;
  const int lane = t & 31, wave = t >> 5;
  const int wm = wave & 3, wn = wave >> 2;
  const int half8 = (lane < 16) ? 0 : 8;
  const int m0 = blockIdx.x * 128, n0 = blockIdx.y * 64;
  v8f acc[2][2] = {};
  for (int k0 = 0; k0 < HH; k0 += 32) {
    for (int i = 0; i < 16; ++i) {               // A tile 128x32
      int idx = t + i * 256;
      lA[idx >> 5][idx & 31] = f2bf(y[(size_t)(m0 + (idx >> 5)) * HH + k0 + (idx & 31)]);
    }
    for (int i = 0; i < 8; ++i) {                // B tile (W1 rows) 64x32
      int idx = t + i * 256;
      lB[idx >> 5][idx & 31] = f2bf(W1[(size_t)(n0 + (idx >> 5)) * HH + k0 + (idx & 31)]);
    }
    __syncthreads();
    v16bf a[2], bb[2];
    for (int mi = 0; mi < 2; ++mi) {
      int row = wm * 32 + mi * 16 + (lane & 15);
      for (int e = 0; e < 16; ++e) a[mi][e] = lA[row][kmap(e, half8)];
    }
    for (int ni = 0; ni < 2; ++ni) {
      int col = wn * 32 + ni * 16 + (lane & 15);
      for (int e = 0; e < 16; ++e) bb[ni][e] = lB[col][kmap(e, half8)];
    }
    for (int mi = 0; mi < 2; ++mi)
      for (int ni = 0; ni < 2; ++ni)
        acc[mi][ni] = __builtin_amdgcn_wmma_f32_16x16x32_bf16(
            false, a[mi], false, bb[ni], (short)0, acc[mi][ni], false, false);
    __syncthreads();
  }
  for (int mi = 0; mi < 2; ++mi)
    for (int ni = 0; ni < 2; ++ni)
      for (int r = 0; r < 8; ++r) {
        int gm = m0 + wm * 32 + mi * 16 + r + half8;
        int gn = n0 + wn * 32 + ni * 16 + (lane & 15);
        yn[(size_t)gm * HH + gn] = f2bf(tanhf(acc[mi][ni][r] + b1[gn]));
      }
}

// ---------------------------------------------------------------------------
// Kernel 2: rowmax[b,d] = max_q A(masked). x row-block (64 x H bf16 = 128KB)
// resident in LDS; y_n tiles (64q x 64k) DMA'd by the TDM, double-buffered.
// Waves: 2(d) x 4(q); 4 WMMAs per wave per stage.
// ---------------------------------------------------------------------------
__global__ __launch_bounds__(256) void rowmax_kernel(
    const float* __restrict__ x, const bf16_t* __restrict__ yn,
    const unsigned char* __restrict__ xmask, const unsigned char* __restrict__ ymask,
    float* __restrict__ rowmax) {
  __shared__ bf16_t lX[64][HH];
  __shared__ bf16_t lY[2][64][64];
  __shared__ float red[4][64];
  const int t = threadIdx.x;
  const int lane = t & 31, wave = t >> 5;
  const int wq = wave & 3, wd = wave >> 2;           // 4 q-strips x 2 d-strips
  const int half8 = (lane < 16) ? 0 : 8;
  const int b = blockIdx.y, d0 = blockIdx.x * 64;
  const bf16_t* ynb = yn + (size_t)b * QQ * HH;
  const unsigned ldsY0 = (unsigned)(uintptr_t)&lY[0][0][0];
  const float* xg = x + ((size_t)b * DD + d0) * HH;
  for (int i = 0; i < 256; ++i) {
    int idx = t + i * 256;
    if (i < 248) __builtin_prefetch(&xg[idx + 2048], 0, 1);
    lX[idx >> 10][idx & 1023] = f2bf(xg[idx]);
  }
  const int NSK = HH / 64;                           // 16 k-stages per q-block
  const int NST = (QQ / 64) * NSK;                   // 128 total stages
  if (wave == 0)                                     // prologue: stage 0
    tdm_load_tile_bf16(ldsY0, ynb, 64, 64, HH, QQ, HH);
  float rm[2][8];
  for (int mi = 0; mi < 2; ++mi)
    for (int r = 0; r < 8; ++r) rm[mi][r] = NEG_INF;
  for (int q0 = 0; q0 < QQ; q0 += 64) {
    v8f acc[2] = {};
    for (int k0 = 0; k0 < HH; k0 += 64) {
      int s = (q0 >> 6) * NSK + (k0 >> 6);
      int buf = s & 1;
      if (wave == 0) __builtin_amdgcn_s_wait_tensorcnt(0);
      __syncthreads();                               // stage s visible to all
      int ns = s + 1;
      if (wave == 0 && ns < NST) {                   // prefetch next stage
        int nq0 = (ns / NSK) * 64, nk0 = (ns % NSK) * 64;
        tdm_load_tile_bf16(ldsY0 + (unsigned)(ns & 1) * 64 * 64 * 2,
                           ynb + (size_t)nq0 * HH + nk0, 64, 64, HH, QQ, HH);
      }
      for (int ks = 0; ks < 64; ks += 32) {
        v16bf bfr;
        int col = wq * 16 + (lane & 15);
        for (int e = 0; e < 16; ++e) bfr[e] = lY[buf][col][ks + kmap(e, half8)];
        for (int mi = 0; mi < 2; ++mi) {
          v16bf a;
          int row = wd * 32 + mi * 16 + (lane & 15);
          for (int e = 0; e < 16; ++e) a[e] = lX[row][k0 + ks + kmap(e, half8)];
          acc[mi] = __builtin_amdgcn_wmma_f32_16x16x32_bf16(
              false, a, false, bfr, (short)0, acc[mi], false, false);
        }
      }
      __syncthreads();                               // done reading buf
    }
    int qc = q0 + wq * 16 + (lane & 15);
    bool ym = ymask[(size_t)b * QQ + qc] != 0;
    if (!ym)
      for (int mi = 0; mi < 2; ++mi)
        for (int r = 0; r < 8; ++r) rm[mi][r] = fmaxf(rm[mi][r], acc[mi][r]);
  }
  for (int mi = 0; mi < 2; ++mi)
    for (int r = 0; r < 8; ++r) {
      float m = rm[mi][r];
      for (int msk = 1; msk <= 8; msk <<= 1) m = fmaxf(m, __shfl_xor(m, msk, 32));
      if ((lane & 15) == 0) red[wq][wd * 32 + mi * 16 + r + half8] = m;
    }
  __syncthreads();
  if (t < 64) {
    int d = d0 + t;
    float m = fmaxf(fmaxf(red[0][t], red[1][t]), fmaxf(red[2][t], red[3][t]));
    if (xmask[(size_t)b * DD + d]) m = NEG_INF;
    rowmax[(size_t)b * DD + d] = m;
  }
}

// ---------------------------------------------------------------------------
// Softmax over D (used for m_alpha and the final output)
// ---------------------------------------------------------------------------
__global__ __launch_bounds__(256) void softmax_d_kernel(
    const float* __restrict__ in, float* __restrict__ out) {
  __shared__ float sred[256];
  const int b = blockIdx.x, t = threadIdx.x;
  const float* row = in + (size_t)b * DD;
  float mx = NEG_INF;
  for (int i = t; i < DD; i += 256) mx = fmaxf(mx, row[i]);
  sred[t] = mx;
  __syncthreads();
  for (int s = 128; s > 0; s >>= 1) {
    if (t < s) sred[t] = fmaxf(sred[t], sred[t + s]);
    __syncthreads();
  }
  mx = sred[0];
  __syncthreads();
  float sum = 0.f;
  for (int i = t; i < DD; i += 256) sum += __expf(row[i] - mx);
  sred[t] = sum;
  __syncthreads();
  for (int s = 128; s > 0; s >>= 1) {
    if (t < s) sred[t] += sred[t + s];
    __syncthreads();
  }
  float inv = 1.f / sred[0];
  for (int i = t; i < DD; i += 256)
    out[(size_t)b * DD + i] = __expf(row[i] - mx) * inv;
}

// m_d[b,h] = sum_d m_alpha[b,d] * x[b,d,h]
__global__ __launch_bounds__(256) void md_kernel(
    const float* __restrict__ x, const float* __restrict__ alpha,
    float* __restrict__ md) {
  __shared__ float la[DD];
  const int b = blockIdx.y, h0 = blockIdx.x * 256, t = threadIdx.x;
  for (int i = t; i < DD; i += 256) la[i] = alpha[(size_t)b * DD + i];
  __syncthreads();
  int h = h0 + t;
  float acc = 0.f;
  const float* xb = x + (size_t)b * DD * HH + h;
  for (int d = 0; d < DD; ++d) acc += la[d] * xb[(size_t)d * HH];
  md[(size_t)b * HH + h] = acc;
}

// v[b,h] = bL[h] + WL[h,:].y1[b,:] + m_d[b,h]   (wave per output row)
__global__ __launch_bounds__(256) void v_kernel(
    const float* __restrict__ y1, const float* __restrict__ WL,
    const float* __restrict__ bL, const float* __restrict__ md,
    float* __restrict__ v) {
  __shared__ float ly[HH];
  const int b = blockIdx.y, t = threadIdx.x;
  const int lane = t & 31, wave = t >> 5;
  for (int i = t; i < HH; i += 256) ly[i] = y1[(size_t)b * HH + i];
  __syncthreads();
  int h = blockIdx.x * 8 + wave;
  float s = 0.f;
  const float* wrow = WL + (size_t)h * HH;
  for (int k = lane; k < HH; k += 32) s += wrow[k] * ly[k];
  for (int m = 1; m <= 16; m <<= 1) s += __shfl_xor(s, m, 32);
  if (lane == 0) v[(size_t)b * HH + h] = s + bL[h] + md[(size_t)b * HH + h];
}

// t[b,q] = y_n[b,q,:].v[b,:]   (wave per row)
__global__ __launch_bounds__(256) void t_kernel(
    const bf16_t* __restrict__ yn, const float* __restrict__ v,
    float* __restrict__ tv) {
  __shared__ float lv[HH];
  const int b = blockIdx.y, t = threadIdx.x;
  const int lane = t & 31, wave = t >> 5;
  for (int i = t; i < HH; i += 256) lv[i] = v[(size_t)b * HH + i];
  __syncthreads();
  int q = blockIdx.x * 8 + wave;
  float s = 0.f;
  const bf16_t* row = yn + ((size_t)b * QQ + q) * HH;
  for (int k = lane; k < HH; k += 32) s += bf2f(row[k]) * lv[k];
  for (int m = 1; m <= 16; m <<= 1) s += __shfl_xor(s, m, 32);
  if (lane == 0) tv[(size_t)b * QQ + q] = s;
}

// ---------------------------------------------------------------------------
// Kernel 7: recompute A tiles (TDM-fed, double-buffered), softmax-over-Q
// numerator/denominator with t[], fold in x.v -> xWy[b,d].
// ---------------------------------------------------------------------------
__global__ __launch_bounds__(256) void final_kernel(
    const float* __restrict__ x, const bf16_t* __restrict__ yn,
    const float* __restrict__ v, const float* __restrict__ tv,
    const float* __restrict__ rowmax,
    const unsigned char* __restrict__ xmask, const unsigned char* __restrict__ ymask,
    float* __restrict__ xWy) {
  __shared__ bf16_t lX[64][HH];
  __shared__ bf16_t lY[2][64][64];
  __shared__ float lV[HH];
  __shared__ float xdv[64];
  __shared__ float tl[64];
  __shared__ float redN[4][64];
  __shared__ float redD[4][64];
  const int t = threadIdx.x;
  const int lane = t & 31, wave = t >> 5;
  const int wq = wave & 3, wd = wave >> 2;
  const int half8 = (lane < 16) ? 0 : 8;
  const int b = blockIdx.y, d0 = blockIdx.x * 64;
  const bf16_t* ynb = yn + (size_t)b * QQ * HH;
  const unsigned ldsY0 = (unsigned)(uintptr_t)&lY[0][0][0];
  const float* xg = x + ((size_t)b * DD + d0) * HH;
  for (int i = 0; i < 256; ++i) {
    int idx = t + i * 256;
    if (i < 248) __builtin_prefetch(&xg[idx + 2048], 0, 1);
    lX[idx >> 10][idx & 1023] = f2bf(xg[idx]);
  }
  for (int i = t; i < HH; i += 256) lV[i] = v[(size_t)b * HH + i];
  const int NSK = HH / 64;
  const int NST = (QQ / 64) * NSK;
  if (wave == 0)
    tdm_load_tile_bf16(ldsY0, ynb, 64, 64, HH, QQ, HH);
  __syncthreads();
  for (int i = 0; i < 8; ++i) {            // x.v per row (wave per 8 rows)
    int row = wave * 8 + i;
    float s = 0.f;
    for (int k = lane; k < HH; k += 32) s += bf2f(lX[row][k]) * lV[k];
    for (int m = 1; m <= 16; m <<= 1) s += __shfl_xor(s, m, 32);
    if (lane == 0) xdv[row] = s;
  }
  float rmaxr[2][8];
  for (int mi = 0; mi < 2; ++mi)
    for (int r = 0; r < 8; ++r)
      rmaxr[mi][r] = rowmax[(size_t)b * DD + d0 + wd * 32 + mi * 16 + r + half8];
  float num[2][8] = {}, den[2][8] = {};
  for (int q0 = 0; q0 < QQ; q0 += 64) {
    __syncthreads();                       // previous tl readers done
    if (t < 64) tl[t] = tv[(size_t)b * QQ + q0 + t];
    v8f acc[2] = {};
    for (int k0 = 0; k0 < HH; k0 += 64) {
      int s = (q0 >> 6) * NSK + (k0 >> 6);
      int buf = s & 1;
      if (wave == 0) __builtin_amdgcn_s_wait_tensorcnt(0);
      __syncthreads();
      int ns = s + 1;
      if (wave == 0 && ns < NST) {
        int nq0 = (ns / NSK) * 64, nk0 = (ns % NSK) * 64;
        tdm_load_tile_bf16(ldsY0 + (unsigned)(ns & 1) * 64 * 64 * 2,
                           ynb + (size_t)nq0 * HH + nk0, 64, 64, HH, QQ, HH);
      }
      for (int ks = 0; ks < 64; ks += 32) {
        v16bf bfr;
        int col = wq * 16 + (lane & 15);
        for (int e = 0; e < 16; ++e) bfr[e] = lY[buf][col][ks + kmap(e, half8)];
        for (int mi = 0; mi < 2; ++mi) {
          v16bf a;
          int row = wd * 32 + mi * 16 + (lane & 15);
          for (int e = 0; e < 16; ++e) a[e] = lX[row][k0 + ks + kmap(e, half8)];
          acc[mi] = __builtin_amdgcn_wmma_f32_16x16x32_bf16(
              false, a, false, bfr, (short)0, acc[mi], false, false);
        }
      }
      __syncthreads();
    }
    int qc = q0 + wq * 16 + (lane & 15);
    bool ym = ymask[(size_t)b * QQ + qc] != 0;
    float tq = tl[qc - q0];
    for (int mi = 0; mi < 2; ++mi)
      for (int r = 0; r < 8; ++r) {
        float e = ym ? 0.f : __expf(acc[mi][r] - rmaxr[mi][r]);
        num[mi][r] += e * tq;
        den[mi][r] += e;
      }
  }
  for (int mi = 0; mi < 2; ++mi)
    for (int r = 0; r < 8; ++r) {
      float n = num[mi][r], dn = den[mi][r];
      for (int m = 1; m <= 8; m <<= 1) {
        n += __shfl_xor(n, m, 32);
        dn += __shfl_xor(dn, m, 32);
      }
      if ((lane & 15) == 0) {
        redN[wq][wd * 32 + mi * 16 + r + half8] = n;
        redD[wq][wd * 32 + mi * 16 + r + half8] = dn;
      }
    }
  __syncthreads();
  if (t < 64) {
    int d = d0 + t;
    float n = redN[0][t] + redN[1][t] + redN[2][t] + redN[3][t];
    float dn = redD[0][t] + redD[1][t] + redD[2][t] + redD[3][t];
    float term = (dn > 0.f) ? (n / dn) : 0.f;
    xWy[(size_t)b * DD + d] =
        xmask[(size_t)b * DD + d] ? NEG_INF : (xdv[t] + term);
  }
}

extern "C" void kernel_launch(void* const* d_in, const int* in_sizes, int n_in,
                              void* d_out, int out_size, void* d_ws, size_t ws_size,
                              hipStream_t stream) {
  (void)in_sizes; (void)n_in; (void)out_size; (void)ws_size;
  const float* x  = (const float*)d_in[0];
  const float* y  = (const float*)d_in[1];
  const float* y1 = (const float*)d_in[2];
  const float* W1 = (const float*)d_in[3];
  const float* b1 = (const float*)d_in[4];
  const float* WL = (const float*)d_in[5];
  const float* bL = (const float*)d_in[6];
  const unsigned char* xmask = (const unsigned char*)d_in[7];
  const unsigned char* ymask = (const unsigned char*)d_in[8];
  float* out = (float*)d_out;

  char* ws = (char*)d_ws;
  bf16_t* yn    = (bf16_t*)ws; ws += (size_t)BB * QQ * HH * 2;  // 32 MB
  float* rowmax = (float*)ws;  ws += (size_t)BB * DD * 4;
  float* malpha = (float*)ws;  ws += (size_t)BB * DD * 4;
  float* md     = (float*)ws;  ws += (size_t)BB * HH * 4;
  float* vv     = (float*)ws;  ws += (size_t)BB * HH * 4;
  float* tvv    = (float*)ws;  ws += (size_t)BB * QQ * 4;
  float* xwy    = (float*)ws;  ws += (size_t)BB * DD * 4;

  yn_gemm_tanh<<<dim3((BB * QQ) / 128, HH / 64), 256, 0, stream>>>(y, W1, b1, yn);
  rowmax_kernel<<<dim3(DD / 64, BB), 256, 0, stream>>>(x, yn, xmask, ymask, rowmax);
  softmax_d_kernel<<<BB, 256, 0, stream>>>(rowmax, malpha);
  md_kernel<<<dim3(HH / 256, BB), 256, 0, stream>>>(x, malpha, md);
  v_kernel<<<dim3(HH / 8, BB), 256, 0, stream>>>(y1, WL, bL, md, vv);
  t_kernel<<<dim3(QQ / 8, BB), 256, 0, stream>>>(yn, vv, tvv);
  final_kernel<<<dim3(DD / 64, BB), 256, 0, stream>>>(x, yn, vv, tvv, rowmax,
                                                      xmask, ymask, xwy);
  softmax_d_kernel<<<BB, 256, 0, stream>>>(xwy, out);
}
